// TopKSAE_28793460752863
// MI455X (gfx1250) — compile-verified, matrix operand order checked by reading, not run on hardware
//
#include <hip/hip_runtime.h>
#include <stdint.h>

#define B_SZ  4096
#define DIN   1024
#define NLAT  16384
#define K_TOP 32

typedef __attribute__((ext_vector_type(16))) __bf16         v16bf;
typedef __attribute__((ext_vector_type(8)))  float          v8f;
typedef __attribute__((ext_vector_type(8)))  unsigned short v8us;
typedef __attribute__((ext_vector_type(16))) unsigned short v16us;

__device__ __forceinline__ unsigned short f32_to_bf16_rne(float f) {
    uint32_t u = __float_as_uint(f);
    uint32_t r = u + 0x7FFFu + ((u >> 16) & 1u);
    return (unsigned short)(r >> 16);
}

// Monotone map: float -> uint32 preserving order (larger float => larger key)
__device__ __forceinline__ uint32_t f32_key(float f) {
    uint32_t u = __float_as_uint(f);
    uint32_t m = (u & 0x80000000u) ? 0xFFFFFFFFu : 0x80000000u;
    return u ^ m;
}

// ---------------------------------------------------------------------------
// Kernel 0: bulk f32 -> bf16 conversion (x and enc_w), float4 -> ushort4.
// ---------------------------------------------------------------------------
__global__ __launch_bounds__(256)
void cvt_f32_bf16(const float* __restrict__ in, unsigned short* __restrict__ out, int n4) {
    const int i = blockIdx.x * 256 + threadIdx.x;
    if (i < n4) {
        float4 f = ((const float4*)in)[i];
        ushort4 o;
        o.x = f32_to_bf16_rne(f.x);
        o.y = f32_to_bf16_rne(f.y);
        o.z = f32_to_bf16_rne(f.z);
        o.w = f32_to_bf16_rne(f.w);
        ((ushort4*)out)[i] = o;
    }
}

// ---------------------------------------------------------------------------
// Kernel 1: dec_w [DIN, NLAT] -> dec_wt [NLAT, DIN] (so decode gathers rows)
// ---------------------------------------------------------------------------
__global__ __launch_bounds__(256)
void sae_transpose_dec(const float* __restrict__ dec_w, float* __restrict__ dec_wt) {
    __shared__ float t[32][33];
    const int tx = threadIdx.x;            // 0..31
    const int ty = threadIdx.y;            // 0..7
    const int l0 = blockIdx.x * 32;
    const int d0 = blockIdx.y * 32;
#pragma unroll
    for (int i = 0; i < 4; ++i)
        t[ty + i * 8][tx] = dec_w[(size_t)(d0 + ty + i * 8) * NLAT + (l0 + tx)];
    __syncthreads();
#pragma unroll
    for (int i = 0; i < 4; ++i)
        dec_wt[(size_t)(l0 + ty + i * 8) * DIN + (d0 + tx)] = t[tx][ty + i * 8];
}

// ---------------------------------------------------------------------------
// Kernel 2: encoder GEMM, bf16 WMMA, f32 accumulate, async global->LDS
// double buffering (ASYNCcnt). pre[b,l] = sum_d x[b,d]*enc_w[l,d] + enc_b[l]
//
// Dynamic LDS layout (byte offsets are true LDS addresses for the async ops):
//   [    0, 10240)  A buf 0   (128 rows x 32 bf16, stride 80 B)
//   [10240, 20480)  A buf 1
//   [20480, 30720)  B buf 0
//   [30720, 40960)  B buf 1
// ---------------------------------------------------------------------------
#define A_ROWB   80          // padded row stride in bytes (40 ushorts)
#define BUF_B    10240       // one tile buffer: 128*80 bytes
#define LDS_TOT  40960

__device__ __forceinline__ void issue_tile(const unsigned short* __restrict__ xb,
                                           const unsigned short* __restrict__ wb,
                                           int tid, int tileM, int tileN, int k0, int p) {
    // 256 threads x 2 x 16B chunks cover each 8 KB tile (2 threads per 64B row)
    const int row = tid >> 1;
    const unsigned c0 = (unsigned)((tid & 1) * 32);
    const unsigned gA = (unsigned)(((tileM + row) * DIN + k0) * 2);
    const unsigned gB = (unsigned)(((tileN + row) * DIN + k0) * 2);
    const unsigned lA = (unsigned)(p * BUF_B + row * A_ROWB);
    const unsigned lB = (unsigned)(2 * BUF_B + p * BUF_B + row * A_ROWB);
#pragma unroll
    for (int i = 0; i < 2; ++i) {
        const unsigned co = c0 + (unsigned)(i * 16);
        unsigned ldsA = lA + co, voffA = gA + co;
        unsigned ldsB = lB + co, voffB = gB + co;
        asm volatile("global_load_async_to_lds_b128 %0, %1, %2"
                     :: "v"(ldsA), "v"(voffA), "s"(xb) : "memory");
        asm volatile("global_load_async_to_lds_b128 %0, %1, %2"
                     :: "v"(ldsB), "v"(voffB), "s"(wb) : "memory");
    }
}

__global__ __launch_bounds__(256)
void sae_encode_gemm(const unsigned short* __restrict__ xb,
                     const unsigned short* __restrict__ wb,
                     const float* __restrict__ enc_b,
                     float* __restrict__ pre) {
    extern __shared__ char dynsm[];

    const int tid   = threadIdx.x;
    const int lane  = tid & 31;
    const int wave  = tid >> 5;                  // 0..7
    const int tileM = blockIdx.y * 128;
    const int tileN = blockIdx.x * 128;

    v8f acc[8];
#pragma unroll
    for (int s = 0; s < 8; ++s)
#pragma unroll
        for (int i = 0; i < 8; ++i) acc[s][i] = 0.0f;

    const int fr = lane & 15;                    // fragment row (M or N)
    const int kb = (lane < 16) ? 0 : 8;          // per-lane K base (ISA 16-bit layout)

    issue_tile(xb, wb, tid, tileM, tileN, 0, 0);

    for (int it = 0; it < DIN / 32; ++it) {
        const int p = it & 1;
        // my async writes into buf p have landed:
        asm volatile("s_wait_asynccnt 0x0" ::: "memory");
        // everyone's writes into buf p visible AND everyone done reading buf p^1:
        __syncthreads();
        if (it + 1 < DIN / 32)
            issue_tile(xb, wb, tid, tileM, tileN, (it + 1) * 32, p ^ 1);

        const unsigned short* aB = (const unsigned short*)dynsm + p * (BUF_B / 2);
        const unsigned short* bB = (const unsigned short*)dynsm + BUF_B + p * (BUF_B / 2);

        // A fragment: lane<16 -> K {0..7,16..23}; lane>=16 -> K {8..15,24..31}
        v8us alo = *(const v8us*)(aB + (wave * 16 + fr) * 40 + kb);
        v8us ahi = *(const v8us*)(aB + (wave * 16 + fr) * 40 + kb + 16);
        v16us av;
#pragma unroll
        for (int i = 0; i < 8; ++i) { av[i] = alo[i]; av[i + 8] = ahi[i]; }
        v16bf afrag = __builtin_bit_cast(v16bf, av);

#pragma unroll
        for (int s = 0; s < 8; ++s) {
            v8us blo = *(const v8us*)(bB + (s * 16 + fr) * 40 + kb);
            v8us bhi = *(const v8us*)(bB + (s * 16 + fr) * 40 + kb + 16);
            v16us bv;
#pragma unroll
            for (int i = 0; i < 8; ++i) { bv[i] = blo[i]; bv[i + 8] = bhi[i]; }
            v16bf bfrag = __builtin_bit_cast(v16bf, bv);
            acc[s] = __builtin_amdgcn_wmma_f32_16x16x32_bf16(
                false, afrag, false, bfrag, (short)0, acc[s], false, false);
        }
        // no trailing barrier needed: next iteration's wait+barrier precedes reuse
    }

    // store C/D (+bias): VGPR r, lane -> (M = r + 8*(lane>>4), N = lane&15)
    const int mh = (lane >> 4) * 8;
#pragma unroll
    for (int s = 0; s < 8; ++s) {
        const int colg = tileN + s * 16 + fr;
        const float bias = enc_b[colg];
#pragma unroll
        for (int r = 0; r < 8; ++r) {
            const int rowg = tileM + wave * 16 + mh + r;
            pre[(size_t)rowg * NLAT + colg] = acc[s][r] + bias;
        }
    }
}

// ---------------------------------------------------------------------------
// Kernel 3: per-row top-32 via 4-pass radix select (row cached in VGPRs),
// wave32 bitonic sort of the 32 winners (descending, low-index tie-break),
// scatter relu'd values into dense acts row, emit sorted idx + vals.
// ---------------------------------------------------------------------------
__global__ __launch_bounds__(256)
void sae_topk(const float* __restrict__ pre, float* __restrict__ acts,
              int* __restrict__ idx_out, float* __restrict__ vals_out) {
    __shared__ unsigned int hist[256];
    __shared__ unsigned int s_prefix, s_remaining, s_nsel, s_eq_taken;
    __shared__ unsigned int s_thr, s_eq_needed;
    __shared__ float l_val[K_TOP];
    __shared__ int   l_idx[K_TOP];

    const int tid = threadIdx.x;
    const int row = blockIdx.x;
    const float* prow = pre + (size_t)row * NLAT;

    float v[64];
#pragma unroll
    for (int c = 0; c < 16; ++c) {
        float4 f = ((const float4*)prow)[c * 256 + tid];
        v[c * 4 + 0] = f.x; v[c * 4 + 1] = f.y;
        v[c * 4 + 2] = f.z; v[c * 4 + 3] = f.w;
    }

    if (tid == 0) { s_prefix = 0; s_remaining = K_TOP; }

    for (int pass = 0; pass < 4; ++pass) {
        const int shift = 24 - pass * 8;
        hist[tid] = 0;                 // blockDim == 256
        __syncthreads();
        const unsigned int pfx = s_prefix;
#pragma unroll
        for (int e = 0; e < 64; ++e) {
            uint32_t u = f32_key(v[e]);
            bool ok = (pass == 0) || ((u >> (shift + 8)) == pfx);
            if (ok) atomicAdd(&hist[(u >> shift) & 255u], 1u);
        }
        __syncthreads();
        if (tid == 0) {
            unsigned int rem = s_remaining, cum = 0;
            int chosen = 0;
            for (int b = 255; b >= 0; --b) {
                unsigned int c2 = cum + hist[b];
                if (c2 >= rem) { chosen = b; s_remaining = rem - cum; break; }
                cum = c2;
            }
            s_prefix = (s_prefix << 8) | (unsigned int)chosen;
        }
        __syncthreads();
    }

    if (tid == 0) {
        s_thr = s_prefix; s_eq_needed = s_remaining;
        s_nsel = 0; s_eq_taken = 0;
    }
    __syncthreads();
    const unsigned int thr = s_thr;
    const unsigned int eqneed = s_eq_needed;

    uint64_t selmask = 0;
#pragma unroll
    for (int e = 0; e < 64; ++e) {
        const uint32_t u = f32_key(v[e]);
        const int col = ((e >> 2) * 256 + tid) * 4 + (e & 3);
        if (u > thr) {
            unsigned int p = atomicAdd(&s_nsel, 1u);
            l_val[p] = v[e]; l_idx[p] = col;
            selmask |= 1ull << e;
        } else if (u == thr) {
            unsigned int t = atomicAdd(&s_eq_taken, 1u);
            if (t < eqneed) {
                unsigned int p = atomicAdd(&s_nsel, 1u);
                l_val[p] = v[e]; l_idx[p] = col;
                selmask |= 1ull << e;
            }
        }
    }

    float* arow = acts + (size_t)row * NLAT;
#pragma unroll
    for (int c = 0; c < 16; ++c) {
        float4 o;
        o.x = (selmask >> (c * 4 + 0)) & 1 ? fmaxf(v[c * 4 + 0], 0.0f) : 0.0f;
        o.y = (selmask >> (c * 4 + 1)) & 1 ? fmaxf(v[c * 4 + 1], 0.0f) : 0.0f;
        o.z = (selmask >> (c * 4 + 2)) & 1 ? fmaxf(v[c * 4 + 2], 0.0f) : 0.0f;
        o.w = (selmask >> (c * 4 + 3)) & 1 ? fmaxf(v[c * 4 + 3], 0.0f) : 0.0f;
        ((float4*)arow)[c * 256 + tid] = o;
    }
    __syncthreads();

    if (tid < 32) {
        float sv = l_val[tid];
        int   si = l_idx[tid];
#pragma unroll
        for (int k = 2; k <= 32; k <<= 1) {
#pragma unroll
            for (int j = k >> 1; j > 0; j >>= 1) {
                float ov = __shfl_xor(sv, j, 32);
                int   oi = __shfl_xor(si, j, 32);
                bool better = (sv > ov) || (sv == ov && si < oi);
                bool dirDesc = ((tid & k) == 0);
                bool keepBetter = dirDesc ? ((tid & j) == 0) : ((tid & j) != 0);
                if (better != keepBetter) { sv = ov; si = oi; }
            }
        }
        idx_out[row * K_TOP + tid]  = si;
        vals_out[row * K_TOP + tid] = fmaxf(sv, 0.0f);
    }
}

// ---------------------------------------------------------------------------
// Kernel 4: sparse decode. recon[b,:] = dec_b + sum_j vals[b,j]*dec_wt[idx[b,j],:]
// dec_wt (67 MB) is L2-resident (192 MB L2) -> gathers hit L2.
// ---------------------------------------------------------------------------
__global__ __launch_bounds__(256)
void sae_decode(const float* __restrict__ vals, const int* __restrict__ idx,
                const float* __restrict__ dec_wt, const float* __restrict__ dec_b,
                float* __restrict__ recon) {
    __shared__ float lv[K_TOP];
    __shared__ int   li[K_TOP];
    const int row = blockIdx.x;
    const int tid = threadIdx.x;
    if (tid < K_TOP) {
        lv[tid] = vals[row * K_TOP + tid];
        li[tid] = idx[row * K_TOP + tid];
    }
    __syncthreads();

    float4 acc = ((const float4*)dec_b)[tid];
#pragma unroll 4
    for (int j = 0; j < K_TOP; ++j) {
        const float s = lv[j];                 // wave-uniform -> scalar branch
        if (s != 0.0f) {
            float4 w = ((const float4*)(dec_wt + (size_t)li[j] * DIN))[tid];
            acc.x += s * w.x; acc.y += s * w.y;
            acc.z += s * w.z; acc.w += s * w.w;
        }
    }
    ((float4*)(recon + (size_t)row * DIN))[tid] = acc;
}

// ---------------------------------------------------------------------------
extern "C" void kernel_launch(void* const* d_in, const int* in_sizes, int n_in,
                              void* d_out, int out_size, void* d_ws, size_t ws_size,
                              hipStream_t stream) {
    (void)in_sizes; (void)n_in; (void)out_size; (void)ws_size;
    const float* x     = (const float*)d_in[0];
    const float* enc_w = (const float*)d_in[1];
    const float* enc_b = (const float*)d_in[2];
    const float* dec_w = (const float*)d_in[3];
    const float* dec_b = (const float*)d_in[4];
    // d_in[5] is k (== 32, compile-time K_TOP)

    float* out   = (float*)d_out;
    float* recon = out;                                          // [B, DIN]
    float* acts  = out + (size_t)B_SZ * DIN;                     // [B, NLAT] (staged pre_acts)
    int*   idx   = (int*)(out + (size_t)B_SZ * DIN + (size_t)B_SZ * NLAT); // [B, 32]

    // Workspace layout (bytes):
    //   [0,          64 MiB)  dec_wt  f32 [NLAT, DIN]
    //   [64 MiB,     +8 MiB)  x_bf16      [B_SZ, DIN]
    //   [72 MiB,    +32 MiB)  w_bf16      [NLAT, DIN]
    //   [104 MiB,  +0.5 MiB)  tvals   f32 [B_SZ, 32]
    char* ws = (char*)d_ws;
    float*          dec_wt = (float*)ws;
    unsigned short* xb     = (unsigned short*)(ws + (size_t)NLAT * DIN * 4);
    unsigned short* wbf    = (unsigned short*)(ws + (size_t)NLAT * DIN * 4 + (size_t)B_SZ * DIN * 2);
    float*          tvals  = (float*)(ws + (size_t)NLAT * DIN * 4 + (size_t)B_SZ * DIN * 2
                                         + (size_t)NLAT * DIN * 2);

    cvt_f32_bf16<<<(B_SZ * DIN / 4 + 255) / 256, 256, 0, stream>>>(x, xb, B_SZ * DIN / 4);
    cvt_f32_bf16<<<(NLAT * DIN / 4 + 255) / 256, 256, 0, stream>>>(enc_w, wbf, NLAT * DIN / 4);
    sae_transpose_dec<<<dim3(NLAT / 32, DIN / 32), dim3(32, 8), 0, stream>>>(dec_w, dec_wt);
    sae_encode_gemm<<<dim3(NLAT / 128, B_SZ / 128), 256, LDS_TOT, stream>>>(xb, wbf, enc_b, acts);
    sae_topk<<<B_SZ, 256, 0, stream>>>(acts, acts, idx, tvals);
    sae_decode<<<B_SZ, 256, 0, stream>>>(tvals, idx, dec_wt, dec_b, recon);
}
